// ClusterNet_16398185136268
// MI455X (gfx1250) — matrix operand order repaired and checked by reading, not compiled, not run on hardware
//
#include <hip/hip_runtime.h>
#include <hip/hip_bf16.h>

// ---------------------------------------------------------------------------
// ClusterNet fused pipeline for MI455X (gfx1250, wave32, WMMA)
// GEMMs in bf16 via v_wmma_f32_16x16x32_bf16, double-buffered LDS staged with
// CDNA5 async global->LDS copies (ASYNCcnt). Sinkhorn via scaling vectors.
// ---------------------------------------------------------------------------

typedef __attribute__((ext_vector_type(16))) __bf16 v16bf;
typedef __attribute__((ext_vector_type(8)))  float  v8f;

union FragBF { v16bf v; unsigned u[8]; };

__device__ __forceinline__ unsigned short f2bf(float x) {
    unsigned u = __float_as_uint(x);
    unsigned r = (u + 0x7FFFu + ((u >> 16) & 1u)) >> 16;   // RNE
    return (unsigned short)r;
}
__device__ __forceinline__ float bf2f(unsigned short h) {
    return __uint_as_float(((unsigned)h) << 16);
}
__device__ __forceinline__ float wave_sum(float x) {
    #pragma unroll
    for (int o = 16; o > 0; o >>= 1) x += __shfl_xor(x, o, 32);
    return x;
}

// CDNA5 async global->LDS copy (ASYNCcnt path). vdst = LDS byte offset,
// vaddr = 64-bit global address, saddr disabled.
__device__ __forceinline__ void async_ld_b128(unsigned ldsoff, const void* g) {
    asm volatile("global_load_async_to_lds_b128 %0, %1, off"
                 :: "v"(ldsoff), "v"(g) : "memory");
}
__device__ __forceinline__ void wait_async0() {
    asm volatile("s_wait_asynccnt 0x0" ::: "memory");
}

// ------------------------------- conversions -------------------------------

__global__ __launch_bounds__(256) void cvt_f32_bf16(const float* __restrict__ in,
                                                    unsigned short* __restrict__ out, int n) {
    int i = blockIdx.x * blockDim.x + threadIdx.x;
    if (i < n) out[i] = f2bf(in[i]);
}

// in [R][C] f32 -> out [C][R] bf16 (K-contiguous B-operand layout)
__global__ __launch_bounds__(256) void tconv_f32_bf16(const float* __restrict__ in,
                                                      unsigned short* __restrict__ out,
                                                      int R, int C) {
    int i = blockIdx.x * blockDim.x + threadIdx.x;
    if (i < R * C) {
        int r = i / C, c = i - r * C;
        out[(size_t)c * R + r] = f2bf(in[i]);
    }
}

// ------------------------------- row norms ---------------------------------

__global__ __launch_bounds__(256) void rownorm_f32(const float* __restrict__ X,
                                                   float* __restrict__ out, int cols) {
    int row  = blockIdx.x * (blockDim.x >> 5) + (threadIdx.x >> 5);
    int lane = threadIdx.x & 31;
    float s = 0.f;
    for (int c = lane; c < cols; c += 32) {
        float v = X[(size_t)row * cols + c];
        s += v * v;
    }
    s = wave_sum(s);
    if (!lane) out[row] = s;
}

__global__ __launch_bounds__(256) void rownorm_bf16(const unsigned short* __restrict__ X,
                                                    float* __restrict__ out, int cols) {
    int row  = blockIdx.x * (blockDim.x >> 5) + (threadIdx.x >> 5);
    int lane = threadIdx.x & 31;
    float s = 0.f;
    for (int c = lane; c < cols; c += 32) {
        float v = bf2f(X[(size_t)row * cols + c]);
        s += v * v;
    }
    s = wave_sum(s);
    if (!lane) out[row] = s;
}

// ------------------------------- WMMA GEMM ---------------------------------
// C[M][N] = A[M][K] @ Bt[N][K]^T, A/Bt bf16 K-contiguous.
// Block tile 128x128, K tile 32, double-buffered LDS, async global->LDS.
// 8 waves, each computes 32x64 (2x4 WMMA tiles).
// MODE 0: relu(acc + bias) -> bf16
// MODE 1: acc + bias       -> bf16
// MODE 2: d = sqrt(max(rowN+colN-2*acc,1e-12)); of1=d; of2=exp(-2d)

#define TILE_M 128
#define TILE_N 128
#define TILE_K 32
#define LDS_STRIDE 40   // ushort units per row: 32 data + 8 pad; 80B = 5*16B (16B aligned)

template <int MODE>
__global__ __launch_bounds__(256) void wmma_gemm(
    const unsigned short* __restrict__ A,
    const unsigned short* __restrict__ Bt,
    int M, int N, int K,
    const float* __restrict__ bias,
    const float* __restrict__ rowN,
    const float* __restrict__ colN,
    unsigned short* __restrict__ obf,
    float* __restrict__ of1,
    float* __restrict__ of2)
{
    __shared__ unsigned short As[2][TILE_M * LDS_STRIDE];
    __shared__ unsigned short Bs[2][TILE_N * LDS_STRIDE];

    const int t     = threadIdx.x;
    const int lane  = t & 31;
    const int wid   = t >> 5;
    const int wm    = wid >> 1;          // 0..3  -> 32-row slab
    const int wn    = wid & 1;           // 0..1  -> 64-col slab
    const int m0    = blockIdx.y * TILE_M;
    const int n0    = blockIdx.x * TILE_N;
    const int khalf = lane >> 4;
    const int lr    = lane & 15;

    // LDS byte offsets of the two buffers (low 32 bits of generic address)
    const unsigned asb[2] = { (unsigned)(unsigned long long)&As[0][0],
                              (unsigned)(unsigned long long)&As[1][0] };
    const unsigned bsb[2] = { (unsigned)(unsigned long long)&Bs[0][0],
                              (unsigned)(unsigned long long)&Bs[1][0] };

    // each thread stages 2 x 16B chunks of A and of Bt per tile
    const int srow0 = t >> 2;            // chunks 0..255 -> rows 0..63
    const int srow1 = (t + 256) >> 2;    // rows 64..127
    const int sc0   = t & 3;             // 16B chunk within 64B row

    v8f acc[2][4];
    #pragma unroll
    for (int i = 0; i < 2; i++)
        #pragma unroll
        for (int j = 0; j < 4; j++)
            #pragma unroll
            for (int e = 0; e < 8; e++) acc[i][j][e] = 0.f;

    int buf = 0;
    // prologue: stage tile kt=0 into buf 0
    async_ld_b128(asb[0] + srow0 * 80 + sc0 * 16, A  + (size_t)(m0 + srow0) * K + sc0 * 8);
    async_ld_b128(asb[0] + srow1 * 80 + sc0 * 16, A  + (size_t)(m0 + srow1) * K + sc0 * 8);
    async_ld_b128(bsb[0] + srow0 * 80 + sc0 * 16, Bt + (size_t)(n0 + srow0) * K + sc0 * 8);
    async_ld_b128(bsb[0] + srow1 * 80 + sc0 * 16, Bt + (size_t)(n0 + srow1) * K + sc0 * 8);

    for (int kt = 0; kt < K; kt += TILE_K) {
        wait_async0();
        __syncthreads();                 // staged tile visible to all waves

        // prefetch next K tile into the other buffer while we compute
        if (kt + TILE_K < K) {
            int nb = buf ^ 1;
            int kn = kt + TILE_K;
            async_ld_b128(asb[nb] + srow0 * 80 + sc0 * 16, A  + (size_t)(m0 + srow0) * K + kn + sc0 * 8);
            async_ld_b128(asb[nb] + srow1 * 80 + sc0 * 16, A  + (size_t)(m0 + srow1) * K + kn + sc0 * 8);
            async_ld_b128(bsb[nb] + srow0 * 80 + sc0 * 16, Bt + (size_t)(n0 + srow0) * K + kn + sc0 * 8);
            async_ld_b128(bsb[nb] + srow1 * 80 + sc0 * 16, Bt + (size_t)(n0 + srow1) * K + kn + sc0 * 8);
        }

        // load fragments per documented 16-bit WMMA VGPR layouts
        const unsigned short* as = &As[buf][0];
        const unsigned short* bs = &Bs[buf][0];
        FragBF fa[2], fb[4];
        #pragma unroll
        for (int mi = 0; mi < 2; mi++) {
            int ro = (wm * 32 + mi * 16 + lr) * LDS_STRIDE;
            #pragma unroll
            for (int v = 0; v < 8; v++) {
                int k = 2 * (v & 3) + 16 * (v >> 2) + 8 * khalf;   // A 16x32 layout
                fa[mi].u[v] = *(const unsigned*)&as[ro + k];
            }
        }
        #pragma unroll
        for (int ni = 0; ni < 4; ni++) {
            int ro = (wn * 64 + ni * 16 + lr) * LDS_STRIDE;
            #pragma unroll
            for (int v = 0; v < 8; v++) {
                int k = 2 * (v & 3) + 16 * (v >> 2) + 8 * khalf;   // B 32x16 layout
                fb[ni].u[v] = *(const unsigned*)&bs[ro + k];
            }
        }

        #pragma unroll
        for (int mi = 0; mi < 2; mi++)
            #pragma unroll
            for (int ni = 0; ni < 4; ni++)
                acc[mi][ni] = __builtin_amdgcn_wmma_f32_16x16x32_bf16(
                    false, fa[mi].v, false, fb[ni].v,
                    (short)0, acc[mi][ni], false, false);

        __syncthreads();                 // all waves done reading buf before re-stage
        buf ^= 1;
    }

    // epilogue: C/D layout -> M = r + 8*khalf, N = lane&15 within each 16x16 tile
    #pragma unroll
    for (int mi = 0; mi < 2; mi++)
        #pragma unroll
        for (int ni = 0; ni < 4; ni++)
            #pragma unroll
            for (int r = 0; r < 8; r++) {
                int grow = m0 + wm * 32 + mi * 16 + khalf * 8 + r;
                int gcol = n0 + wn * 64 + ni * 16 + lr;
                float val = acc[mi][ni][r];
                size_t oi = (size_t)grow * N + gcol;
                if (MODE == 0) {
                    float v = fmaxf(val + bias[gcol], 0.f);
                    obf[oi] = f2bf(v);
                } else if (MODE == 1) {
                    obf[oi] = f2bf(val + bias[gcol]);
                } else {
                    float d2 = rowN[grow] + colN[gcol] - 2.f * val;
                    float d  = sqrtf(fmaxf(d2, 1e-12f));
                    of1[oi] = d;
                    of2[oi] = __expf(-2.0f * d);   // exp(-d/eps), eps=0.5
                }
            }
}

// ------------------------------- sinkhorn ----------------------------------
// E fixed; Q = E ∘ alpha ∘ beta.  alpha[k] = (1/K)/Σ_b E*beta ;  beta[b] = (1/B)/Σ_k E*alpha

__global__ __launch_bounds__(256) void fill1_k(float* p, int n) {
    int i = blockIdx.x * blockDim.x + threadIdx.x;
    if (i < n) p[i] = 1.f;
}

__global__ __launch_bounds__(256) void col_alpha_k(const float* __restrict__ E,
                                                   const float* __restrict__ beta,
                                                   float* __restrict__ alpha, int B, int K) {
    int k = blockIdx.x * blockDim.x + threadIdx.x;   // K exact multiple of 256
    float acc = 0.f;
    for (int b = 0; b < B; b++) acc += E[(size_t)b * K + k] * beta[b];
    alpha[k] = (1.0f / (float)K) / acc;
}

__global__ __launch_bounds__(256) void row_beta_k(const float* __restrict__ E,
                                                  const float* __restrict__ alpha,
                                                  float* __restrict__ beta, int B, int K) {
    int b    = blockIdx.x * (blockDim.x >> 5) + (threadIdx.x >> 5);
    int lane = threadIdx.x & 31;
    float s = 0.f;
    for (int k = lane; k < K; k += 32) s += E[(size_t)b * K + k] * alpha[k];
    s = wave_sum(s);
    if (!lane) beta[b] = (1.0f / (float)B) / s;
}

// out layout: [0]=loss, [1..B]=assign, [1+B..)=soft_counts(K), raw(K), cc(K)
__global__ __launch_bounds__(256) void init_out_k(const int* __restrict__ raw,
                                                  const int* __restrict__ cc,
                                                  float* __restrict__ out, int B, int K) {
    int i = blockIdx.x * blockDim.x + threadIdx.x;
    if (i == 0) out[0] = 0.f;
    if (i < K) {
        out[1 + B + K + i]     = (float)raw[i];
        out[1 + B + 2 * K + i] = (float)cc[i];
    }
}

__global__ __launch_bounds__(256) void finalize_k(const float* __restrict__ dist,
                                                  const float* __restrict__ E,
                                                  const float* __restrict__ alpha,
                                                  float* __restrict__ out,
                                                  float* __restrict__ tinv, int B, int K) {
    int b    = blockIdx.x * (blockDim.x >> 5) + (threadIdx.x >> 5);
    int lane = threadIdx.x & 31;
    float minD = 3.4e38f; int aD = 0;    // argmin of dist       -> raw counts
    float minS = 3.4e38f; int aS = 0;    // argmin of soft ∝ E*α -> assignments
    float s = 0.f;
    for (int k = lane; k < K; k += 32) {
        float d  = dist[(size_t)b * K + k];
        float ea = E[(size_t)b * K + k] * alpha[k];
        s += ea;
        if (d  < minD) { minD = d;  aD = k; }
        if (ea < minS) { minS = ea; aS = k; }
    }
    s = wave_sum(s);
    #pragma unroll
    for (int o = 16; o > 0; o >>= 1) {
        float od = __shfl_xor(minD, o, 32); int oi = __shfl_xor(aD, o, 32);
        if (od < minD || (od == minD && oi < aD)) { minD = od; aD = oi; }
        float os = __shfl_xor(minS, o, 32); int oj = __shfl_xor(aS, o, 32);
        if (os < minS || (os == minS && oj < aS)) { minS = os; aS = oj; }
    }
    if (!lane) {
        tinv[b]    = 1.0f / s;
        out[1 + b] = (float)aS;
        float dpick = dist[(size_t)b * K + aS];
        atomicAdd(&out[0], dpick * (1.0f / (float)B));
        atomicAdd(&out[1 + B + K + aD], 1.0f);
        atomicAdd(&out[1 + B + 2 * K + aS], 1.0f);
    }
}

__global__ __launch_bounds__(256) void soft_counts_k(const float* __restrict__ E,
                                                     const float* __restrict__ alpha,
                                                     const float* __restrict__ tinv,
                                                     float* __restrict__ out, int B, int K) {
    int k = blockIdx.x * blockDim.x + threadIdx.x;
    float acc = 0.f;
    for (int b = 0; b < B; b++) acc += E[(size_t)b * K + k] * tinv[b];
    out[1 + B + k] = alpha[k] * acc;
}

// ------------------------------- launcher ----------------------------------

extern "C" void kernel_launch(void* const* d_in, const int* in_sizes, int n_in,
                              void* d_out, int out_size, void* d_ws, size_t ws_size,
                              hipStream_t stream) {
    const int B = 8192, DIN = 768, H = 4096, NZ = 512, NC = 1024;

    const float* inp  = (const float*)d_in[0];
    const float* W1   = (const float*)d_in[1];
    const float* b1   = (const float*)d_in[2];
    const float* W2   = (const float*)d_in[3];
    const float* b2   = (const float*)d_in[4];
    const float* cent = (const float*)d_in[5];
    const int*   raw  = (const int*)d_in[6];
    const int*   cc   = (const int*)d_in[7];
    float* out = (float*)d_out;

    // workspace carve-up (all sizes multiples of 256B)
    char* w = (char*)d_ws;
    size_t o = 0;
    unsigned short* inp_bf  = (unsigned short*)(w + o); o += (size_t)B * DIN * 2;   // 12.6 MB
    unsigned short* W1t     = (unsigned short*)(w + o); o += (size_t)H * DIN * 2;   //  6.3 MB
    unsigned short* W2t     = (unsigned short*)(w + o); o += (size_t)NZ * H * 2;    //  4.2 MB
    unsigned short* cent_bf = (unsigned short*)(w + o); o += (size_t)NC * NZ * 2;   //  1.0 MB
    unsigned short* h_bf    = (unsigned short*)(w + o); o += (size_t)B * H * 2;     // 67.1 MB
    unsigned short* f_bf    = (unsigned short*)(w + o); o += (size_t)B * NZ * 2;    //  8.4 MB
    float* fn    = (float*)(w + o); o += (size_t)B * 4;
    float* cn    = (float*)(w + o); o += (size_t)NC * 4;
    float* dist  = (float*)(w + o); o += (size_t)B * NC * 4;                        // 33.6 MB
    float* Emat  = (float*)(w + o); o += (size_t)B * NC * 4;                        // 33.6 MB
    float* alpha = (float*)(w + o); o += (size_t)NC * 4;
    float* beta  = (float*)(w + o); o += (size_t)B * 4;
    float* tinv  = (float*)(w + o); o += (size_t)B * 4;

    const int T = 256;

    // 1) precision conversions / transposes
    cvt_f32_bf16<<<(B * DIN + T - 1) / T, T, 0, stream>>>(inp, inp_bf, B * DIN);
    tconv_f32_bf16<<<(DIN * H + T - 1) / T, T, 0, stream>>>(W1, W1t, DIN, H);
    tconv_f32_bf16<<<(H * NZ + T - 1) / T, T, 0, stream>>>(W2, W2t, H, NZ);
    cvt_f32_bf16<<<(NC * NZ + T - 1) / T, T, 0, stream>>>(cent, cent_bf, NC * NZ);
    rownorm_f32<<<NC / 8, T, 0, stream>>>(cent, cn, NZ);

    // 2) GEMM chain (WMMA bf16, async-LDS double buffered)
    wmma_gemm<0><<<dim3(H / TILE_N, B / TILE_M), T, 0, stream>>>(
        inp_bf, W1t, B, H, DIN, b1, nullptr, nullptr, h_bf, nullptr, nullptr);
    wmma_gemm<1><<<dim3(NZ / TILE_N, B / TILE_M), T, 0, stream>>>(
        h_bf, W2t, B, NZ, H, b2, nullptr, nullptr, f_bf, nullptr, nullptr);
    rownorm_bf16<<<B / 8, T, 0, stream>>>(f_bf, fn, NZ);
    wmma_gemm<2><<<dim3(NC / TILE_N, B / TILE_M), T, 0, stream>>>(
        f_bf, cent_bf, B, NC, NZ, nullptr, fn, cn, nullptr, dist, Emat);

    // 3) Sinkhorn on scaling vectors (15 iters)
    fill1_k<<<(B + T - 1) / T, T, 0, stream>>>(beta, B);
    init_out_k<<<(NC + T - 1) / T, T, 0, stream>>>(raw, cc, out, B, NC);
    for (int it = 0; it < 15; it++) {
        col_alpha_k<<<NC / T, T, 0, stream>>>(Emat, beta, alpha, B, NC);
        row_beta_k<<<B / 8, T, 0, stream>>>(Emat, alpha, beta, B, NC);
    }

    // 4) outputs: loss, assignments, counts
    finalize_k<<<B / 8, T, 0, stream>>>(dist, Emat, alpha, out, tinv, B, NC);
    soft_counts_k<<<NC / T, T, 0, stream>>>(Emat, alpha, tinv, out, B, NC);

    (void)in_sizes; (void)n_in; (void)out_size; (void)ws_size;
}
